// PointTransformerBlock_661424963758
// MI455X (gfx1250) — matrix operand order repaired and checked by reading, not compiled, not run on hardware
//
#include <hip/hip_runtime.h>
#include <hip/hip_bf16.h>

// ---------------------------------------------------------------------------
// Point Transformer block for MI455X (gfx1250, wave32, WMMA).
// knn_kernel -> proj_kernel (q/k/v WMMA GEMMs) -> fused pt_attn_kernel.
// One wave per query: M=16 neighbors, K=64 (2x k=32 WMMA), N=64 (4 tiles).
// Weights are staged in LDS PRE-SWIZZLED into B-fragment order so every
// fragment load is a contiguous 32B LDS read (ds_load_b128) instead of
// 16 scalar ds_load_u16.
// ---------------------------------------------------------------------------

typedef __attribute__((ext_vector_type(16))) _Float16 v16h;
typedef __attribute__((ext_vector_type(8)))  _Float16 v8h;
typedef __attribute__((ext_vector_type(8)))  float    v8f;
typedef __attribute__((ext_vector_type(4)))  float    v4f;

constexpr int B_  = 4;
constexpr int N_  = 4096;
constexpr int DIM = 64;
constexpr int KNN = 16;
constexpr int TS  = 72;   // f16 LDS tile row stride (16B-aligned rows: 144B)

__device__ __forceinline__ v8f wmma_f32_16x16x32_f16(v16h a, v16h b, v8f c) {
  // (neg_a, A, neg_b, B, c_mod, C, reuse_a, reuse_b)
  return __builtin_amdgcn_wmma_f32_16x16x32_f16(false, a, false, b, (short)0, c,
                                                false, false);
}

// A-fragment K mapping (16-bit A 16x32, ISA 7.12.2):
// lanes 0-15: VGPR0-3 -> K 0..7, VGPR4-7 -> K 16..23 ; lanes 16-31: +8
__device__ __forceinline__ int kmapA(int e, int hi) {
  return (e < 8) ? (8 * hi + e) : (16 + 8 * hi + (e - 8));
}

// ---------------------------------------------------------------------------
// Stage one 64x64 f32 weight into LDS in B-fragment order:
//   dst[(ks*4+n)*32 + lane] = v16h with e -> W[(ks*32 + kmapA(e,hi))*64 + 16n+lo]
// (B layout matches A: lanes 0-15 hold K 0..7,16..23, lanes 16-31 hold K 8..15,24..31, col = lane&15)
// Requires exactly 256 threads: one fragment per thread.
// ---------------------------------------------------------------------------
__device__ __forceinline__ void stage_weight_frags(const float* __restrict__ W,
                                                   v16h* __restrict__ dst,
                                                   int tid) {
  const int fi = tid >> 5, lane = tid & 31;
  const int ks = fi >> 2, n = fi & 3;
  const int hi = lane >> 4, lo = lane & 15;
  v16h f;
#pragma unroll
  for (int e = 0; e < 16; ++e) {
    int K = ks * 32 + kmapA(e, hi);
    f[e] = (_Float16)W[K * 64 + 16 * n + lo];
  }
  dst[fi * 32 + lane] = f;   // 32B contiguous -> ds_store_b128 x2
}

// Load A fragment from a 16xTS f16 tile row (two contiguous 16B runs).
__device__ __forceinline__ v16h load_a_tile(const _Float16* __restrict__ row,
                                            int ks, int hi) {
  v8h x0 = *(const v8h*)(row + ks * 32 + 8 * hi);
  v8h x1 = *(const v8h*)(row + ks * 32 + 16 + 8 * hi);
  return __builtin_shufflevector(x0, x1, 0, 1, 2, 3, 4, 5, 6, 7, 8, 9, 10, 11,
                                 12, 13, 14, 15);
}

// ---------------------------------------------------------------------------
// Kernel 1: brute-force kNN (stable top-16 by (dist, index), matches argsort)
// ---------------------------------------------------------------------------
__global__ __launch_bounds__(256) void knn_kernel(const float* __restrict__ pos,
                                                  int* __restrict__ knn) {
  __shared__ float sp[256 * 3];
  const int b  = blockIdx.x / (N_ / 256);
  const int qi = (blockIdx.x % (N_ / 256)) * 256 + threadIdx.x;
  const float* pb = pos + (size_t)b * N_ * 3;

  const float qx = pb[qi * 3 + 0], qy = pb[qi * 3 + 1], qz = pb[qi * 3 + 2];

  float bd[KNN];
  int   bi[KNN];
#pragma unroll
  for (int t = 0; t < KNN; ++t) { bd[t] = 3.4e38f; bi[t] = N_; }

  for (int tile = 0; tile < N_; tile += 256) {
    __syncthreads();
#pragma unroll
    for (int t = 0; t < 3; ++t)
      sp[threadIdx.x + 256 * t] = pb[tile * 3 + threadIdx.x + 256 * t];
    __syncthreads();

    for (int j = 0; j < 256; ++j) {
      float dx = qx - sp[j * 3 + 0];
      float dy = qy - sp[j * 3 + 1];
      float dz = qz - sp[j * 3 + 2];
      float d  = dx * dx + dy * dy + dz * dz;
      if (d < bd[KNN - 1]) {            // strict: stable tie-break (arg order)
        int p = KNN - 1;
#pragma unroll
        for (int t = KNN - 2; t >= 0; --t)
          if (d < bd[t]) p = t;
#pragma unroll
        for (int t = KNN - 1; t > 0; --t) {
          if (t > p) { bd[t] = bd[t - 1]; bi[t] = bi[t - 1]; }
        }
#pragma unroll
        for (int t = 0; t < KNN; ++t)
          if (t == p) { bd[t] = d; bi[t] = tile + j; }
      }
    }
  }

  int* o = knn + ((size_t)b * N_ + qi) * KNN;
#pragma unroll
  for (int t = 0; t < KNN; ++t) o[t] = bi[t];
}

// ---------------------------------------------------------------------------
// Kernel 2: q/k/v projections  [16384,64] @ [64,64] x3, WMMA f16->f32
// One wave computes a 16-row x 64-col strip; 8 waves/block -> 128 rows/block.
// ---------------------------------------------------------------------------
__global__ __launch_bounds__(256) void proj_kernel(
    const float* __restrict__ x, const float* __restrict__ wq,
    const float* __restrict__ wk, const float* __restrict__ wv,
    float* __restrict__ qo, float* __restrict__ ko, float* __restrict__ vo) {
  __shared__ v16h sW[3][8 * 32];     // fragment-ordered weights, 8KB each
  const int tid = threadIdx.x;

  stage_weight_frags(wq, sW[0], tid);
  stage_weight_frags(wk, sW[1], tid);
  stage_weight_frags(wv, sW[2], tid);
  __syncthreads();

  const int wave = tid >> 5, lane = tid & 31;
  const int hi = lane >> 4, lo = lane & 15;
  const int row0 = blockIdx.x * 128 + wave * 16;

  // A fragments (shared by all three weights); runs of 8 floats -> b128 loads
  const float* xr = x + (size_t)(row0 + lo) * 64;
  v16h a[2];
#pragma unroll
  for (int ks = 0; ks < 2; ++ks) {
#pragma unroll
    for (int run = 0; run < 2; ++run) {
      int koff = ks * 32 + run * 16 + 8 * hi;
      v4f f0 = *(const v4f*)(xr + koff);
      v4f f1 = *(const v4f*)(xr + koff + 4);
#pragma unroll
      for (int e = 0; e < 4; ++e) {
        a[ks][run * 8 + e]     = (_Float16)f0[e];
        a[ks][run * 8 + 4 + e] = (_Float16)f1[e];
      }
    }
  }

  float* outs[3] = {qo, ko, vo};
#pragma unroll
  for (int w = 0; w < 3; ++w) {
#pragma unroll
    for (int n = 0; n < 4; ++n) {
      v8f c = {};
      c = wmma_f32_16x16x32_f16(a[0], sW[w][(0 * 4 + n) * 32 + lane], c);
      c = wmma_f32_16x16x32_f16(a[1], sW[w][(1 * 4 + n) * 32 + lane], c);
#pragma unroll
      for (int r = 0; r < 8; ++r)
        outs[w][(size_t)(row0 + r + hi * 8) * 64 + 16 * n + lo] = c[r];
    }
  }
}

// ---------------------------------------------------------------------------
// Kernel 3: fused attention.  One wave per query point (8 queries / block).
// ---------------------------------------------------------------------------
__global__ __launch_bounds__(256) void pt_attn_kernel(
    const float* __restrict__ x, const float* __restrict__ pos,
    const int* __restrict__ knn, const float* __restrict__ qb,
    const float* __restrict__ kb, const float* __restrict__ vb,
    const float* __restrict__ pw1, const float* __restrict__ pb1,
    const float* __restrict__ pw2, const float* __restrict__ pb2,
    const float* __restrict__ aw1, const float* __restrict__ ab1,
    const float* __restrict__ aw2, const float* __restrict__ ab2,
    const float* __restrict__ fw, const float* __restrict__ fb,
    float* __restrict__ out) {
  __shared__ v16h sPW2[8 * 32], sAW1[8 * 32], sAW2[8 * 32];  // frag order
  __shared__ float sPW1[3 * 64];
  __shared__ float sPB1[64], sPB2[64], sAB1[64], sAB2[64];
  __shared__ _Float16 sTile[8][16 * TS];  // per-wave D->A relayout tile (f16)
  __shared__ float sQ[8][64];
  __shared__ float sRel[8][16 * 4];
  __shared__ int   sIdx[8][16];
  __shared__ float sAgg[8][64];

  const int tid = threadIdx.x, wave = tid >> 5, lane = tid & 31;
  const int hi = lane >> 4, lo = lane & 15;

  // Stage fragment-ordered f16 weights + small f32 params
  stage_weight_frags(pw2, sPW2, tid);
  stage_weight_frags(aw1, sAW1, tid);
  stage_weight_frags(aw2, sAW2, tid);
  for (int i = tid; i < 3 * 64; i += 256) sPW1[i] = pw1[i];
  if (tid < 64) {
    sPB1[tid] = pb1[tid]; sPB2[tid] = pb2[tid];
    sAB1[tid] = ab1[tid]; sAB2[tid] = ab2[tid];
  }

  const int    gq = blockIdx.x * 8 + wave;   // global query row (b*N + i)
  const int    b  = gq / N_, qi = gq % N_;
  const size_t rowbase = (size_t)b * N_;
  const int*   myknn   = knn + (size_t)gq * KNN;

  if (lane < KNN) {
    int nb = myknn[lane];
    sIdx[wave][lane] = nb;
    const float* pq = pos + (rowbase + qi) * 3;
    const float* pn = pos + (rowbase + nb) * 3;
    sRel[wave][lane * 4 + 0] = pq[0] - pn[0];
    sRel[wave][lane * 4 + 1] = pq[1] - pn[1];
    sRel[wave][lane * 4 + 2] = pq[2] - pn[2];
  }
#pragma unroll
  for (int t = 0; t < 2; ++t)
    sQ[wave][lane + 32 * t] = qb[(rowbase + qi) * 64 + lane + 32 * t];
  __syncthreads();

  // ---- pos_enc: h = relu(rel @ pw1 + pb1) built directly in A layout; ----
  // ---- pe = h @ pw2 + pb2 via WMMA, kept in D fragments.               ----
  const float r0 = sRel[wave][lo * 4 + 0];
  const float r1 = sRel[wave][lo * 4 + 1];
  const float r2 = sRel[wave][lo * 4 + 2];

  v8f pe[4] = {};
#pragma unroll
  for (int ks = 0; ks < 2; ++ks) {
    v16h a;
#pragma unroll
    for (int run = 0; run < 2; ++run) {
      int c0 = ks * 32 + run * 16 + 8 * hi;   // 8 contiguous output channels
      v4f wx0 = *(const v4f*)(sPW1 + 0 * 64 + c0);
      v4f wx1 = *(const v4f*)(sPW1 + 0 * 64 + c0 + 4);
      v4f wy0 = *(const v4f*)(sPW1 + 1 * 64 + c0);
      v4f wy1 = *(const v4f*)(sPW1 + 1 * 64 + c0 + 4);
      v4f wz0 = *(const v4f*)(sPW1 + 2 * 64 + c0);
      v4f wz1 = *(const v4f*)(sPW1 + 2 * 64 + c0 + 4);
      v4f bb0 = *(const v4f*)(sPB1 + c0);
      v4f bb1 = *(const v4f*)(sPB1 + c0 + 4);
#pragma unroll
      for (int e = 0; e < 4; ++e) {
        float h0 = bb0[e] + r0 * wx0[e] + r1 * wy0[e] + r2 * wz0[e];
        float h1 = bb1[e] + r0 * wx1[e] + r1 * wy1[e] + r2 * wz1[e];
        a[run * 8 + e]     = (_Float16)fmaxf(h0, 0.f);
        a[run * 8 + 4 + e] = (_Float16)fmaxf(h1, 0.f);
      }
    }
#pragma unroll
    for (int n = 0; n < 4; ++n)
      pe[n] = wmma_f32_16x16x32_f16(a, sPW2[(ks * 4 + n) * 32 + lane], pe[n]);
  }
#pragma unroll
  for (int n = 0; n < 4; ++n) {
    float bias = sPB2[16 * n + lo];
#pragma unroll
    for (int r = 0; r < 8; ++r) pe[n][r] += bias;
  }
  // pe tile to LDS (f16) for A-layout access in attn layer 1
#pragma unroll
  for (int n = 0; n < 4; ++n)
#pragma unroll
    for (int r = 0; r < 8; ++r)
      sTile[wave][(r + hi * 8) * TS + 16 * n + lo] = (_Float16)pe[n][r];
  __syncthreads();

  // ---- attn layer 1: t1 = relu((q - kg + pe) @ aw1 + ab1) ----
  const int    nbrA = sIdx[wave][lo];                 // A row = neighbor 'lo'
  const float* kr   = kb + (rowbase + nbrA) * 64;
  const _Float16* tileRow = &sTile[wave][lo * TS];
  const float*    qw      = &sQ[wave][0];
  v8f t1[4] = {};
#pragma unroll
  for (int ks = 0; ks < 2; ++ks) {
    v16h a;
#pragma unroll
    for (int run = 0; run < 2; ++run) {
      int koff = ks * 32 + run * 16 + 8 * hi;
      v8h tv  = *(const v8h*)(tileRow + koff);
      v4f qa0 = *(const v4f*)(qw + koff);
      v4f qa1 = *(const v4f*)(qw + koff + 4);
      v4f ka0 = *(const v4f*)(kr + koff);
      v4f ka1 = *(const v4f*)(kr + koff + 4);
#pragma unroll
      for (int e = 0; e < 4; ++e) {
        a[run * 8 + e] = (_Float16)((float)tv[e] + qa0[e] - ka0[e]);
        a[run * 8 + 4 + e] = (_Float16)((float)tv[4 + e] + qa1[e] - ka1[e]);
      }
    }
#pragma unroll
    for (int n = 0; n < 4; ++n)
      t1[n] = wmma_f32_16x16x32_f16(a, sAW1[(ks * 4 + n) * 32 + lane], t1[n]);
  }
  __syncthreads();
#pragma unroll
  for (int n = 0; n < 4; ++n) {
    float bias = sAB1[16 * n + lo];
#pragma unroll
    for (int r = 0; r < 8; ++r)
      sTile[wave][(r + hi * 8) * TS + 16 * n + lo] =
          (_Float16)fmaxf(t1[n][r] + bias, 0.f);
  }
  __syncthreads();

  // ---- attn layer 2: at = t1 @ aw2 + ab2 ----
  v8f at[4] = {};
#pragma unroll
  for (int ks = 0; ks < 2; ++ks) {
    v16h a = load_a_tile(tileRow, ks, hi);
#pragma unroll
    for (int n = 0; n < 4; ++n)
      at[n] = wmma_f32_16x16x32_f16(a, sAW2[(ks * 4 + n) * 32 + lane], at[n]);
  }

  // ---- softmax over k (16 rows split across lane pair l / l^16) + agg ----
  float outacc[4];
#pragma unroll
  for (int n = 0; n < 4; ++n) {
    float bias = sAB2[16 * n + lo];
    float s[8];
    float m = -3.4e38f;
#pragma unroll
    for (int r = 0; r < 8; ++r) {
      s[r] = (at[n][r] + bias) * 0.125f;     // / sqrt(dim)
      m = fmaxf(m, s[r]);
    }
    m = fmaxf(m, __shfl_xor(m, 16, 32));
    float sum = 0.f;
#pragma unroll
    for (int r = 0; r < 8; ++r) { s[r] = __expf(s[r] - m); sum += s[r]; }
    sum += __shfl_xor(sum, 16, 32);
    float inv = 1.f / sum;

    float acc = 0.f;
#pragma unroll
    for (int r = 0; r < 8; ++r) {
      int   nb = sIdx[wave][r + hi * 8];
      float vv = vb[(rowbase + nb) * 64 + 16 * n + lo] + pe[n][r];
      acc += s[r] * inv * vv;
    }
    acc += __shfl_xor(acc, 16, 32);
    outacc[n] = acc;
  }
  if (hi == 0) {
#pragma unroll
    for (int n = 0; n < 4; ++n) sAgg[wave][16 * n + lo] = outacc[n];
  }
  __syncthreads();

  // ---- final: out = agg @ fw + fb + x  (2 columns per lane, VALU) ----
#pragma unroll
  for (int t = 0; t < 2; ++t) {
    int   c = lane + 32 * t;
    float o = fb[c] + x[(rowbase + qi) * 64 + c];
#pragma unroll 8
    for (int u = 0; u < 64; ++u) o += sAgg[wave][u] * fw[u * 64 + c];
    out[(rowbase + qi) * 64 + c] = o;
  }
}

// ---------------------------------------------------------------------------
extern "C" void kernel_launch(void* const* d_in, const int* in_sizes, int n_in,
                              void* d_out, int out_size, void* d_ws,
                              size_t ws_size, hipStream_t stream) {
  (void)in_sizes; (void)n_in; (void)out_size; (void)ws_size;

  const float* x   = (const float*)d_in[0];
  const float* pos = (const float*)d_in[1];
  // d_in[2] = k (scalar, fixed 16)
  const float* wq  = (const float*)d_in[3];
  const float* wk  = (const float*)d_in[4];
  const float* wv  = (const float*)d_in[5];
  const float* pw1 = (const float*)d_in[6];
  const float* pb1 = (const float*)d_in[7];
  const float* pw2 = (const float*)d_in[8];
  const float* pb2 = (const float*)d_in[9];
  const float* aw1 = (const float*)d_in[10];
  const float* ab1 = (const float*)d_in[11];
  const float* aw2 = (const float*)d_in[12];
  const float* ab2 = (const float*)d_in[13];
  const float* fw  = (const float*)d_in[14];
  const float* fb  = (const float*)d_in[15];
  float* out = (float*)d_out;

  const size_t R = (size_t)B_ * N_;            // 16384 rows
  float* qbuf = (float*)d_ws;                  // R*64 f32
  float* kbuf = qbuf + R * DIM;                // R*64 f32
  float* vbuf = kbuf + R * DIM;                // R*64 f32
  int*   knn  = (int*)(vbuf + R * DIM);        // R*16 i32

  knn_kernel<<<dim3((unsigned)(R / 256)), dim3(256), 0, stream>>>(pos, knn);
  proj_kernel<<<dim3((unsigned)(R / 128)), dim3(256), 0, stream>>>(
      x, wq, wk, wv, qbuf, kbuf, vbuf);
  pt_attn_kernel<<<dim3((unsigned)(R / 8)), dim3(256), 0, stream>>>(
      x, pos, knn, qbuf, kbuf, vbuf, pw1, pb1, pw2, pb2, aw1, ab1, aw2, ab2,
      fw, fb, out);
}
